// AdditiveAttention_27238682591788
// MI455X (gfx1250) — matrix operand order repaired and checked by reading, not compiled, not run on hardware
//
#include <hip/hip_runtime.h>
#include <math.h>

// ---------------------------------------------------------------------------
// AdditiveAttention on gfx1250 (MI455X, wave32, WMMA).
//   1) QP = queries @ W_q^T   (bf16 WMMA, f32 accum)   2048x256
//   2) KP = keys    @ W_k^T   (bf16 WMMA, f32 accum)   4096x256
//   3) attn = softmax_k(mask(sum_h w_v[h]*tanh(QP+KP)))  -- v_tanh_f32 bound,
//      8 query rows per block so each KP float4 is reused 8x (L2 traffic /8)
//   4) out = attn @ values    (bf16 WMMA, f32 accum)
// ---------------------------------------------------------------------------

typedef __attribute__((ext_vector_type(16))) __bf16 v16bf;
typedef __attribute__((ext_vector_type(8)))  float  v8f;

__device__ __forceinline__ float fast_tanh(float x) {
#if __has_builtin(__builtin_amdgcn_tanhf)
  return __builtin_amdgcn_tanhf(x);      // v_tanh_f32 (gfx1250 TRANS unit)
#else
  return tanhf(x);
#endif
}

__device__ __forceinline__ float fast_exp(float x) {
#if __has_builtin(__builtin_amdgcn_exp2f)
  return __builtin_amdgcn_exp2f(x * 1.4426950408889634f);  // v_exp_f32
#else
  return expf(x);
#endif
}

__device__ __forceinline__ float wave_max(float v) {
#pragma unroll
  for (int off = 16; off > 0; off >>= 1) v = fmaxf(v, __shfl_xor(v, off, 32));
  return v;
}
__device__ __forceinline__ float wave_sum(float v) {
#pragma unroll
  for (int off = 16; off > 0; off >>= 1) v += __shfl_xor(v, off, 32);
  return v;
}

// ---------------------------------------------------------------------------
// Y[M,H] = X[M,D] @ W[H,D]^T    one 16x16 tile per wave, K-loop of 32.
// A-frag (16x32 bf16):  half=lane>>4, m=lane&15
//   VGPR i<4 : K = 8*half + 2i,{+1};  VGPR i>=4 : K = 16 + 8*half + 2(i-4),{+1}
// B-frag (32x16 bf16):  n = lane&15, K same as A-frag; B[k,n]=W[n,k]
// C/D (16x16 f32):      row = 8*half + r, col = lane&15
// ---------------------------------------------------------------------------
__global__ __launch_bounds__(256)
void aa_gemm_xWt(const float* __restrict__ X, const float* __restrict__ W,
                 float* __restrict__ Y, int M, int D, int H) {
  const int wave   = (blockIdx.x * blockDim.x + threadIdx.x) >> 5;
  const int lane   = threadIdx.x & 31;
  const int tilesN = H >> 4;
  const int tm = wave / tilesN;
  const int tn = wave % tilesN;
  if (tm * 16 >= M) return;                 // wave-uniform guard (EXEC stays full)
  const int half = lane >> 4;
  const int r16  = lane & 15;

  const float* __restrict__ xrow = X + (size_t)(tm * 16 + r16) * D;
  const float* __restrict__ wrow = W + (size_t)(tn * 16 + r16) * D;

  v8f acc = {};
  for (int kb = 0; kb < D; kb += 32) {
    v16bf a, b;
#pragma unroll
    for (int i = 0; i < 8; ++i) {
      const int ka  = (i < 4) ? (half * 8 + 2 * i) : (16 + half * 8 + 2 * (i - 4));
      const int kbb = (i < 4) ? (half * 8 + 2 * i) : (16 + half * 8 + 2 * (i - 4));
      a[2 * i]     = (__bf16)xrow[kb + ka];
      a[2 * i + 1] = (__bf16)xrow[kb + ka + 1];
      b[2 * i]     = (__bf16)wrow[kb + kbb];
      b[2 * i + 1] = (__bf16)wrow[kb + kbb + 1];
    }
    acc = __builtin_amdgcn_wmma_f32_16x16x32_bf16(
        /*neg_a=*/false, a, /*neg_b=*/false, b,
        /*c_mod=*/(short)0, acc, /*reuse_a=*/false, /*reuse_b=*/false);
  }
#pragma unroll
  for (int r = 0; r < 8; ++r)
    Y[(size_t)(tm * 16 + half * 8 + r) * H + tn * 16 + r16] = acc[r];
}

// ---------------------------------------------------------------------------
// Scores + masked softmax. Block = (b, q-tile of QT=8 rows), 256 threads.
// Thread tid owns key k=tid for all 8 query rows; KP row streamed once per
// block (float4 + prefetch), reused 8x from registers; sQ tile broadcast
// from LDS. Softmax: wave32 shuffle reduce + 8-entry cross-wave partials.
// ---------------------------------------------------------------------------
#define QT 8
__global__ __launch_bounds__(256)
void aa_scores_softmax(const float* __restrict__ QP, const float* __restrict__ KP,
                       const float* __restrict__ wv, const int* __restrict__ valid_lens,
                       float* __restrict__ attn) {
  const int H = 256, Nk = 256, Nq = 128;
  __shared__ float sQ[QT][256];
  __shared__ float sW[256];
  __shared__ float partM[8];
  __shared__ float partS[8];

  const int b  = blockIdx.y;
  const int q0 = blockIdx.x * QT;
  const int tid  = threadIdx.x;
  const int lane = tid & 31;
  const int wid  = tid >> 5;

  sW[tid] = wv[tid];
#pragma unroll
  for (int qq = 0; qq < QT; ++qq)
    sQ[qq][tid] = QP[(size_t)(b * Nq + q0 + qq) * H + tid];
  __syncthreads();

  const float* __restrict__ krow = KP + (size_t)(b * Nk + tid) * H;

  float s[QT];
#pragma unroll
  for (int qq = 0; qq < QT; ++qq) s[qq] = 0.0f;

  for (int h = 0; h < H; h += 32) {                  // one 128B cacheline / lane
    __builtin_prefetch(krow + h + 64, 0, 0);         // global_prefetch_b8, 2 lines ahead
#pragma unroll
    for (int hh = h; hh < h + 32; hh += 4) {
      const float4 kv = *(const float4*)(krow + hh); // 16B aligned (H=256)
#pragma unroll
      for (int j = 0; j < 4; ++j) {
        const float kvj = (j == 0) ? kv.x : (j == 1) ? kv.y : (j == 2) ? kv.z : kv.w;
        const float wj  = sW[hh + j];                // uniform LDS broadcast
#pragma unroll
        for (int qq = 0; qq < QT; ++qq)
          s[qq] = fmaf(wj, fast_tanh(sQ[qq][hh + j] + kvj), s[qq]);
      }
    }
  }

  const int valid = valid_lens[b];
#pragma unroll
  for (int qq = 0; qq < QT; ++qq)
    s[qq] = (tid < valid) ? s[qq] : -1.0e6f;

  // Per-row masked softmax over k = 0..255 (one value per thread).
  for (int qq = 0; qq < QT; ++qq) {
    const float wm = wave_max(s[qq]);
    if (lane == 0) partM[wid] = wm;
    __syncthreads();
    float bm = partM[0];
#pragma unroll
    for (int i = 1; i < 8; ++i) bm = fmaxf(bm, partM[i]);

    const float e  = fast_exp(s[qq] - bm);
    const float ws = wave_sum(e);
    if (lane == 0) partS[wid] = ws;
    __syncthreads();
    float bs = partS[0];
#pragma unroll
    for (int i = 1; i < 8; ++i) bs += partS[i];

    attn[(size_t)(b * Nq + q0 + qq) * Nk + tid] = e * (1.0f / bs);
    __syncthreads();   // protect partM/partS before next row
  }
}

// ---------------------------------------------------------------------------
// out[b] = attn[b] (128x256) @ values[b] (256x256), bf16 WMMA, f32 accum.
// ---------------------------------------------------------------------------
__global__ __launch_bounds__(256)
void aa_gemm_av(const float* __restrict__ A, const float* __restrict__ Vv,
                float* __restrict__ Y) {
  const int M = 128, K = 256, N = 256;
  const int wave = (blockIdx.x * blockDim.x + threadIdx.x) >> 5;
  const int lane = threadIdx.x & 31;
  const int tilesPerBatch = (M / 16) * (N / 16);  // 128
  const int b  = wave / tilesPerBatch;
  const int t  = wave % tilesPerBatch;
  const int tm = t / (N / 16);
  const int tn = t % (N / 16);
  const int half = lane >> 4;
  const int r16  = lane & 15;

  const float* __restrict__ arow = A  + (size_t)(b * M + tm * 16 + r16) * K;
  const float* __restrict__ vb   = Vv + (size_t)b * K * N;
  const int n = tn * 16 + r16;

  v8f acc = {};
  for (int kb = 0; kb < K; kb += 32) {
    v16bf af, bfv;
#pragma unroll
    for (int i = 0; i < 8; ++i) {
      const int ka = (i < 4) ? (half * 8 + 2 * i) : (16 + half * 8 + 2 * (i - 4));
      af[2 * i]     = (__bf16)arow[kb + ka];
      af[2 * i + 1] = (__bf16)arow[kb + ka + 1];
      const int kk = kb + ka;
      bfv[2 * i]     = (__bf16)vb[(size_t)kk * N + n];
      bfv[2 * i + 1] = (__bf16)vb[(size_t)(kk + 1) * N + n];
    }
    acc = __builtin_amdgcn_wmma_f32_16x16x32_bf16(
        false, af, false, bfv, (short)0, acc, false, false);
  }
#pragma unroll
  for (int r = 0; r < 8; ++r)
    Y[(size_t)(b * M + tm * 16 + half * 8 + r) * N + tn * 16 + r16] = acc[r];
}

// ---------------------------------------------------------------------------
extern "C" void kernel_launch(void* const* d_in, const int* in_sizes, int n_in,
                              void* d_out, int out_size, void* d_ws, size_t ws_size,
                              hipStream_t stream) {
  (void)in_sizes; (void)n_in; (void)out_size; (void)ws_size;
  const float* queries = (const float*)d_in[0];   // (16,128,256)
  const float* keys    = (const float*)d_in[1];   // (16,256,256)
  const float* values  = (const float*)d_in[2];   // (16,256,256)
  const int*   valid   = (const int*)  d_in[3];   // (16,)
  const float* W_q     = (const float*)d_in[4];   // (256,256)
  const float* W_k     = (const float*)d_in[5];   // (256,256)
  const float* w_v     = (const float*)d_in[6];   // (256,)
  float* out = (float*)d_out;                     // (16,128,256)

  const int B = 16, Nq = 128, Nk = 256, D = 256, H = 256, V = 256;

  float* QP = (float*)d_ws;                 // 2048*256 f32 = 2 MB
  float* KP = QP + (size_t)B * Nq * H;      // 4096*256 f32 = 4 MB
  float* AT = KP + (size_t)B * Nk * H;      // 2048*256 f32 = 2 MB

  {
    const int tiles = (B * Nq / 16) * (H / 16);     // 2048 waves
    aa_gemm_xWt<<<tiles / 8, 256, 0, stream>>>(queries, W_q, QP, B * Nq, D, H);
  }
  {
    const int tiles = (B * Nk / 16) * (H / 16);     // 4096 waves
    aa_gemm_xWt<<<tiles / 8, 256, 0, stream>>>(keys, W_k, KP, B * Nk, D, H);
  }

  aa_scores_softmax<<<dim3(Nq / QT, B), 256, 0, stream>>>(QP, KP, w_v, valid, AT);

  {
    const int tiles = B * (Nq / 16) * (V / 16);     // 2048 waves
    aa_gemm_av<<<tiles / 8, 256, 0, stream>>>(AT, values, out);
  }
}
